// MinConv2dGRUCell_35287451304714
// MI455X (gfx1250) — matrix operand order repaired and checked by reading, not compiled, not run on hardware
//
#include <hip/hip_runtime.h>
#include <hip/hip_bf16.h>

// ---------------------------------------------------------------------------
// Fused conv3x3(64->128) + minGRU log-space scan for MI455X (gfx1250).
// Conv via 3xBF16-split V_WMMA_F32_16X16X32_BF16; frame t+1 staged with
// GLOBAL_LOAD_ASYNC_TO_LDS_B128 (ASYNCcnt) underneath frame t's WMMAs.
// ---------------------------------------------------------------------------

typedef __bf16 bf16;
typedef bf16  v16bf __attribute__((ext_vector_type(16)));
typedef bf16  v8bf  __attribute__((ext_vector_type(8)));
typedef float v8f   __attribute__((ext_vector_type(8)));

#define B_    4
#define S_    32
#define CIN   64
#define HID   64
#define H_    32
#define W_    32
#define KK    576          // 9 * 64 contraction, order k = (kh*3+kw)*64 + ic
#define WP    584          // padded K pitch for weight LDS (bank rotation, 16B rows)
#define ICP   72           // padded ic pitch for x LDS (144B stride -> conflict-free)
#define YP    65           // padded pixel pitch for conv-result LDS
#define OUT_ELEMS (B_*S_*HID*H_*W_)   // 8388608

// Async copy of one 16B chunk: global -> LDS, tracked by ASYNCcnt.
__device__ __forceinline__ void async_ld_b128(unsigned lds_off, const void* gaddr) {
    asm volatile("global_load_async_to_lds_b128 %0, %1, off"
                 :: "v"(lds_off), "v"(gaddr) : "memory");
}
__device__ __forceinline__ void wait_async0() {
    asm volatile("s_wait_asynccnt 0x0" ::: "memory");
}

__global__ __launch_bounds__(128)
void fused_conv_mingru(const float* __restrict__ x,
                       const float* __restrict__ h0,
                       const float* __restrict__ Wt,
                       const float* __restrict__ bias,
                       float* __restrict__ out)
{
    __shared__ alignas(16) bf16  ws_hi[32 * WP];        // 37376 B
    __shared__ alignas(16) bf16  ws_lo[32 * WP];        // 37376 B
    __shared__ alignas(16) bf16  xs_hi[3 * 34 * ICP];   // 14688 B
    __shared__ alignas(16) bf16  xs_lo[3 * 34 * ICP];   // 14688 B
    __shared__ alignas(16) float ys[32 * YP];           //  8320 B
    __shared__ alignas(16) float xraw[3 * CIN * 32];    // 24576 B  (raw fp32 frame tile)

    const int tid = threadIdx.x;
    const int bx  = blockIdx.x;          // 512 blocks
    const int r0  = bx & 31;             // output image row
    const int cb  = (bx >> 5) & 3;       // 16-channel scan block
    const int b   = bx >> 7;             // batch

    const unsigned xraw_base = (unsigned)(unsigned long long)(const void*)xraw;

    // ---- stage weight slice once: rows m<16 -> gate oc, m>=16 -> hidden oc.
    for (int idx = tid; idx < 32 * KK; idx += 128) {
        int m  = idx / KK;
        int k  = idx - m * KK;
        int khkw = k >> 6;
        int ic   = k & 63;
        int kh = khkw / 3, kw = khkw - kh * 3;
        int oc = (m < 16) ? (cb * 16 + m) : (64 + cb * 16 + (m - 16));
        float w = Wt[((oc * CIN + ic) * 3 + kh) * 3 + kw];
        bf16 hi = (bf16)w;
        ws_hi[m * WP + k] = hi;
        ws_lo[m * WP + k] = (bf16)(w - (float)hi);
    }

    // ---- kick off async staging of frame 0 (in-bounds rows only; 16B chunks).
    {
        const float* x0 = x + (size_t)((b * S_) * CIN) * 1024;
#pragma unroll
        for (int it = 0; it < 12; ++it) {
            int q  = tid + it * 128;     // 0..1535 chunks
            int r  = q >> 9;
            int ic = (q >> 3) & 63;
            int cw = q & 7;
            int gr = r0 - 1 + r;
            if ((unsigned)gr < 32u)
                async_ld_b128(xraw_base + (unsigned)((((r * 64 + ic) * 32) + cw * 4) * 4),
                              x0 + ic * 1024 + gr * 32 + cw * 4);
        }
    }

    // ---- per-thread scan state: channel c (0..15), 4 pixels.
    const int c  = tid & 15;
    const int g  = tid >> 4;
    const int p0 = g * 4;
    const int cg = cb * 16 + c;
    const float bg = bias[cg];
    const float bh = bias[64 + cg];
    float lh[4];
#pragma unroll
    for (int i = 0; i < 4; ++i)
        lh[i] = logf(h0[(b * HID + cg) * 1024 + r0 * 32 + p0 + i]);

    // ---- wave/lane decomposition for the 32x32 GEMM tile (4 waves, 1 tile each).
    const int lane = tid & 31;
    const int wave = tid >> 5;
    const int mt = wave & 1;
    const int nt = wave >> 1;
    const int lm = lane & 15;
    const int lg = lane >> 4;

    __syncthreads();                     // weights visible

    for (int t = 0; t < S_; ++t) {
        // ---- wait for this frame's async tile, then make it block-visible.
        wait_async0();
        __syncthreads();

        // ---- convert raw tile -> bf16 hi/lo with zero halo (rows/cols pad).
        for (int idx = tid; idx < 3 * CIN * 34; idx += 128) {
            int r   = idx / (CIN * 34);
            int rem = idx - r * (CIN * 34);
            int ic  = rem / 34;
            int cc  = rem - ic * 34;
            int gr  = r0 - 1 + r;
            int gc  = cc - 1;
            float v = 0.f;
            if ((unsigned)gr < 32u && (unsigned)gc < 32u)
                v = xraw[(r * 64 + ic) * 32 + gc];
            bf16 hi = (bf16)v;
            int off = (r * 34 + cc) * ICP + ic;
            xs_hi[off] = hi;
            xs_lo[off] = (bf16)(v - (float)hi);
        }
        __syncthreads();                 // xs ready; xraw fully consumed

        // ---- issue async staging of frame t+1; runs under the WMMAs + scan.
        if (t + 1 < S_) {
            const float* xn = x + (size_t)((b * S_ + t + 1) * CIN) * 1024;
#pragma unroll
            for (int it = 0; it < 12; ++it) {
                int q  = tid + it * 128;
                int r  = q >> 9;
                int ic = (q >> 3) & 63;
                int cw = q & 7;
                int gr = r0 - 1 + r;
                if ((unsigned)gr < 32u)
                    async_ld_b128(xraw_base + (unsigned)((((r * 64 + ic) * 32) + cw * 4) * 4),
                                  xn + ic * 1024 + gr * 32 + cw * 4);
            }
        }

        // ---- implicit GEMM: D(32oc x 32pix) = W(32 x 576) * patches(576 x 32).
        v8f acc = {0.f, 0.f, 0.f, 0.f, 0.f, 0.f, 0.f, 0.f};
#pragma unroll
        for (int s = 0; s < 18; ++s) {
            const int khkw = s >> 1;
            const int kh  = khkw / 3;
            const int kw2 = khkw - kh * 3;
            const int ich = (s & 1) * 32;

            // A fragment: per-lane run [klow..+15].
            const int klow = lg * 16;
            const bf16* pa  = &ws_hi[(mt * 16 + lm) * WP + s * 32 + klow];
            const bf16* pal = &ws_lo[(mt * 16 + lm) * WP + s * 32 + klow];
            v8bf a0 = *(const v8bf*)pa;
            v8bf a1 = *(const v8bf*)(pa + 8);
            v8bf l0 = *(const v8bf*)pal;
            v8bf l1 = *(const v8bf*)(pal + 8);
            v16bf ahi = __builtin_shufflevector(a0, a1, 0,1,2,3,4,5,6,7,8,9,10,11,12,13,14,15);
            v16bf alo = __builtin_shufflevector(l0, l1, 0,1,2,3,4,5,6,7,8,9,10,11,12,13,14,15);

            // B fragment: column = pixel, 16 consecutive ic per lane.
            const int pcol = nt * 16 + lm;
            const int boff = (kh * 34 + pcol + kw2) * ICP + ich + lg * 16;
            const bf16* pb  = &xs_hi[boff];
            const bf16* pbl = &xs_lo[boff];
            v8bf b0 = *(const v8bf*)pb;
            v8bf b1 = *(const v8bf*)(pb + 8);
            v8bf m0 = *(const v8bf*)pbl;
            v8bf m1 = *(const v8bf*)(pbl + 8);
            v16bf bhi = __builtin_shufflevector(b0, b1, 0,1,2,3,4,5,6,7,8,9,10,11,12,13,14,15);
            v16bf blo = __builtin_shufflevector(m0, m1, 0,1,2,3,4,5,6,7,8,9,10,11,12,13,14,15);

            // 3xBF16 split-precision accumulation (hi*hi + hi*lo + lo*hi).
            acc = __builtin_amdgcn_wmma_f32_16x16x32_bf16(false, ahi, false, bhi, (short)0, acc, false, false);
            acc = __builtin_amdgcn_wmma_f32_16x16x32_bf16(false, ahi, false, blo, (short)0, acc, false, false);
            acc = __builtin_amdgcn_wmma_f32_16x16x32_bf16(false, alo, false, bhi, (short)0, acc, false, false);
        }

        // ---- scatter C tile: VGPR r -> row r + 8*lanegroup, col = lane%16.
#pragma unroll
        for (int r = 0; r < 8; ++r) {
            int m = mt * 16 + r + 8 * lg;
            int p = nt * 16 + lm;
            ys[m * YP + p] = acc[r];
        }
        __syncthreads();

        // ---- minGRU log-space scan update + output store.
#pragma unroll
        for (int i = 0; i < 4; ++i) {
            int p = p0 + i;
            float gte = ys[c * YP + p] + bg;          // gate pre-activation
            float hd  = ys[(16 + c) * YP + p] + bh;   // hidden pre-activation
            float sp  = log1pf(expf(-fabsf(gte)));
            float log_coeff = -(fmaxf(gte, 0.f) + sp);          // -softplus(g)
            float log_z     =  fminf(gte, 0.f) - sp;            // -softplus(-g)
            float log_th = (hd >= 0.f) ? logf(hd + 0.5f)
                                       : (hd - log1pf(expf(hd)));  // log_g
            float aa = log_coeff + lh[i];
            float bb = log_z + log_th;
            float mx = fmaxf(aa, bb);
            lh[i] = mx + log1pf(expf(-fabsf(aa - bb)));          // logaddexp
            float o = expf(lh[i]);
            size_t oidx = ((size_t)((b * S_ + t) * HID + cg) << 10)
                        + (unsigned)(r0 * 32 + p);
            out[oidx] = o;
            if (t == S_ - 1)
                out[(size_t)OUT_ELEMS + ((size_t)(b * HID + cg) << 10)
                    + (unsigned)(r0 * 32 + p)] = o;              // h_next
        }
        // no trailing barrier: next iteration's wait_async0+__syncthreads orders
        // scan reads of ys/xs before any overwrite.
    }
}

extern "C" void kernel_launch(void* const* d_in, const int* in_sizes, int n_in,
                              void* d_out, int out_size, void* d_ws, size_t ws_size,
                              hipStream_t stream) {
    const float* x  = (const float*)d_in[0];
    const float* h0 = (const float*)d_in[1];
    const float* W  = (const float*)d_in[2];
    const float* bs = (const float*)d_in[3];
    float* out = (float*)d_out;
    // grid: 4 batches x 4 channel-blocks x 32 rows = 512 WGs of 128 threads (4 waves)
    hipLaunchKernelGGL(fused_conv_mingru, dim3(512), dim3(128), 0, stream,
                       x, h0, W, bs, out);
}
